// WindowAttention_86904368267856
// MI455X (gfx1250) — matrix operand order repaired and checked by reading, not compiled, not run on hardware
//
#include <hip/hip_runtime.h>
#include <hip/hip_bf16.h>

typedef __attribute__((ext_vector_type(16))) _Float16 v16h;
typedef __attribute__((ext_vector_type(8)))  _Float16 v8h;
typedef __attribute__((ext_vector_type(8)))  float    v8f;

// ---------------------------------------------------------------------------
// Fragment loaders for V_WMMA_F32_16X16X32_F16 (wave32), per ISA 7.12.2:
//  A (16x32, MxK): lane L holds row M=L%16;  VGPR0-3 : K = kb + (L/16)*8 + 0..7
//                                            VGPR4-7 : K = kb + 16 + (L/16)*8 + 0..7
//  B (32x16, KxN): lane L holds col N=L%16;  VGPR0-7 : same K order as A
// Both are exactly two 16-byte contiguous loads per lane.
// ---------------------------------------------------------------------------
__device__ __forceinline__ v16h load_a_frag(const _Float16* p, int ld, int mr,
                                            int kb, int lane) {
  const int row = mr + (lane & 15);
  const int hf  = lane >> 4;
  const v8h lo = *(const v8h*)(p + row * ld + kb + hf * 8);
  const v8h hi = *(const v8h*)(p + row * ld + kb + 16 + hf * 8);
  v16h r;
#pragma unroll
  for (int i = 0; i < 8; ++i) { r[i] = lo[i]; r[i + 8] = hi[i]; }
  return r;
}

// B fragment for C = A * W^T where W is row-major [n][k] (weights, K-matrix,
// or V^T): element B[k][n] = W[n][k].
__device__ __forceinline__ v16h load_bt_frag(const _Float16* p, int ld, int nb,
                                             int kb, int lane) {
  const int row = nb + (lane & 15);
  const int hf  = lane >> 4;
  const v8h lo = *(const v8h*)(p + row * ld + kb + hf * 8);
  const v8h hi = *(const v8h*)(p + row * ld + kb + 16 + hf * 8);
  v16h r;
#pragma unroll
  for (int i = 0; i < 8; ++i) { r[i] = lo[i]; r[i + 8] = hi[i]; }
  return r;
}

__device__ __forceinline__ v8f wmma_f16(v16h a, v16h b, v8f c) {
  return __builtin_amdgcn_wmma_f32_16x16x32_f16(false, a, false, b, (short)0, c,
                                                false, false);
}

// ---------------------------------------------------------------------------
// Prep: weights fp32 -> f16 (L2-resident, shared by all 8192 blocks) and
// pre-gathered relative-position bias  biasg[h][n][m] = rpb[rpi[n][m]][h].
// ---------------------------------------------------------------------------
__global__ __launch_bounds__(256) void wa_prep_kernel(
    const float* __restrict__ wq, const float* __restrict__ wkv,
    const float* __restrict__ wp, const float* __restrict__ rpb,
    const int* __restrict__ rpi, _Float16* __restrict__ wq16,
    _Float16* __restrict__ wkv16, _Float16* __restrict__ wp16,
    float* __restrict__ biasg) {
  const int i = blockIdx.x * 256 + threadIdx.x;
  if (i < 65536)  wq16[i]  = (_Float16)wq[i];
  if (i < 131072) wkv16[i] = (_Float16)wkv[i];
  if (i < 65536)  wp16[i]  = (_Float16)wp[i];
  if (i < 32768) {                      // 8 heads * 64 * 64
    const int h  = i >> 12;
    const int nm = i & 4095;
    biasg[i] = rpb[rpi[nm] * 8 + h];
  }
}

// ---------------------------------------------------------------------------
// Main kernel: one 256-thread block (8 wave32) per window.
// 2x2 register blocking: one A-fragment pair + one B-fragment pair -> 4 WMMA.
// ---------------------------------------------------------------------------
__global__ __launch_bounds__(256) void wa_attn_kernel(
    const float* __restrict__ x, const float* __restrict__ mask,
    const float* __restrict__ bq, const float* __restrict__ bkv,
    const float* __restrict__ bp, const _Float16* __restrict__ wq16,
    const _Float16* __restrict__ wkv16, const _Float16* __restrict__ wp16,
    const float* __restrict__ biasg, float* __restrict__ out, int nW) {
  __shared__ __align__(16) _Float16 xs[64 * 256];   // x (f16); later reused as O
  __shared__ __align__(16) _Float16 qs[64 * 256];   // Q * scale
  __shared__ __align__(16) _Float16 ks[64 * 256];   // K
  __shared__ __align__(16) _Float16 vt[256 * 64];   // V transposed [feat][tok]
  __shared__ __align__(16) float    sbuf[64 * 64];  // attention logits (f32)
  __shared__ __align__(16) _Float16 pbuf[64 * 64];  // softmax probs (f16)

  const int tid  = threadIdx.x;
  const int lane = tid & 31;
  const int wv   = tid >> 5;
  const int hf   = lane >> 4;
  const int ln   = lane & 15;
  const int blk  = blockIdx.x;
  const float scale = 0.17677669529663687f;  // 1/sqrt(32)

  // ---- Phase 1: stream x window [64][256] fp32 -> f16 LDS ----
  const float4* xg = (const float4*)(x + (size_t)blk * 64 * 256);
#pragma unroll
  for (int i = tid; i < 4096; i += 256) {
    const float4 v = xg[i];
    _Float16* d = xs + i * 4;
    d[0] = (_Float16)v.x; d[1] = (_Float16)v.y;
    d[2] = (_Float16)v.z; d[3] = (_Float16)v.w;
  }
  __syncthreads();

  // ---- Phase 2: Q / K / V projections. 192 16x16 tiles = 48 2x2 quads ----
  for (int t = wv; t < 48; t += 8) {
    const int which = t >> 4;            // 0=Q 1=K 2=V
    const int idx   = t & 15;
    const int mq = idx >> 3, nq = idx & 7;  // m-pair (0..1), n-pair (0..7)
    const int fb = nq * 32;
    const _Float16* W = (which == 0) ? wq16 : wkv16;
    const int wrow = (which == 2) ? (256 + fb) : fb;
    v8f acc[2][2] = {};
#pragma unroll 4
    for (int kk = 0; kk < 256; kk += 32) {
      const v16h a0 = load_a_frag(xs, 256, mq * 32, kk, lane);
      const v16h a1 = load_a_frag(xs, 256, mq * 32 + 16, kk, lane);
      const v16h b0 = load_bt_frag(W, 256, wrow, kk, lane);
      const v16h b1 = load_bt_frag(W, 256, wrow + 16, kk, lane);
      acc[0][0] = wmma_f16(a0, b0, acc[0][0]);
      acc[0][1] = wmma_f16(a0, b1, acc[0][1]);
      acc[1][0] = wmma_f16(a1, b0, acc[1][0]);
      acc[1][1] = wmma_f16(a1, b1, acc[1][1]);
    }
#pragma unroll
    for (int j = 0; j < 2; ++j) {
      const int n = fb + j * 16 + ln;
      if (which == 0) {
        const float bv = bq[n];
#pragma unroll
        for (int i = 0; i < 2; ++i)
#pragma unroll
          for (int r = 0; r < 8; ++r) {
            const int m = mq * 32 + i * 16 + r + hf * 8;
            qs[m * 256 + n] = (_Float16)((acc[i][j][r] + bv) * scale);
          }
      } else if (which == 1) {
        const float bv = bkv[n];
#pragma unroll
        for (int i = 0; i < 2; ++i)
#pragma unroll
          for (int r = 0; r < 8; ++r) {
            const int m = mq * 32 + i * 16 + r + hf * 8;
            ks[m * 256 + n] = (_Float16)(acc[i][j][r] + bv);
          }
      } else {
        const float bv = bkv[256 + n];
#pragma unroll
        for (int i = 0; i < 2; ++i)
#pragma unroll
          for (int r = 0; r < 8; ++r) {  // V stored transposed for P@V
            const int m = mq * 32 + i * 16 + r + hf * 8;
            vt[n * 64 + m] = (_Float16)(acc[i][j][r] + bv);
          }
      }
    }
  }
  __syncthreads();

  // ---- Phase 3: per-head attention ----
  const float* maskw = mask + (size_t)(blk % nW) * 4096;
  for (int h = 0; h < 8; ++h) {
    const float* biash = biasg + h * 4096;
    // S = Q_h K_h^T + bias + mask : 16 tiles = 8 (1m x 2n) pairs, one per wave
    {
      const int mt = wv >> 1;
      const int nb = (wv & 1) * 32;
      const v16h a  = load_a_frag(qs, 256, mt * 16, h * 32, lane);
      const v16h b0 = load_bt_frag(ks, 256, nb, h * 32, lane);
      const v16h b1 = load_bt_frag(ks, 256, nb + 16, h * 32, lane);
      v8f acc0 = {}, acc1 = {};
      acc0 = wmma_f16(a, b0, acc0);
      acc1 = wmma_f16(a, b1, acc1);
#pragma unroll
      for (int j = 0; j < 2; ++j) {
        const v8f& acc = j ? acc1 : acc0;
        const int n = nb + j * 16 + ln;
#pragma unroll
        for (int r = 0; r < 8; ++r) {
          const int m = mt * 16 + r + hf * 8;
          sbuf[m * 64 + n] = acc[r] + biash[m * 64 + n] + maskw[m * 64 + n];
        }
      }
    }
    __syncthreads();
    // Softmax: 4 lanes per row, shfl_xor reduction within the quad (wave32)
    {
      const int row = tid >> 2, part = tid & 3;
      const float* sp = sbuf + row * 64 + part * 16;
      float v[16];
#pragma unroll
      for (int j = 0; j < 16; ++j) v[j] = sp[j];
      float mx = v[0];
#pragma unroll
      for (int j = 1; j < 16; ++j) mx = fmaxf(mx, v[j]);
      mx = fmaxf(mx, __shfl_xor(mx, 1, 32));
      mx = fmaxf(mx, __shfl_xor(mx, 2, 32));
      float sum = 0.f;
#pragma unroll
      for (int j = 0; j < 16; ++j) { v[j] = __expf(v[j] - mx); sum += v[j]; }
      sum += __shfl_xor(sum, 1, 32);
      sum += __shfl_xor(sum, 2, 32);
      const float inv = 1.0f / sum;
      unsigned* pp = (unsigned*)(pbuf + row * 64 + part * 16);
#pragma unroll
      for (int j = 0; j < 8; ++j) {
        _Float16 e0 = (_Float16)(v[2 * j] * inv);
        _Float16 e1 = (_Float16)(v[2 * j + 1] * inv);
        unsigned u = (unsigned)*(unsigned short*)&e0 |
                     ((unsigned)*(unsigned short*)&e1 << 16);
        pp[j] = u;
      }
    }
    __syncthreads();
    // O_h = P V_h : 8 tiles (one per wave), K=64 -> 2 WMMA
    {
      const int mt = wv >> 1, nt = wv & 1;
      v8f acc = {};
#pragma unroll
      for (int kk = 0; kk < 64; kk += 32) {
        const v16h a  = load_a_frag(pbuf, 64, mt * 16, kk, lane);
        const v16h bf = load_bt_frag(vt, 64, h * 32 + nt * 16, kk, lane);
        acc = wmma_f16(a, bf, acc);
      }
      const int n = h * 32 + nt * 16 + ln;
#pragma unroll
      for (int r = 0; r < 8; ++r) {     // xs reused as the O buffer
        const int m = mt * 16 + r + hf * 8;
        xs[m * 256 + n] = (_Float16)acc[r];
      }
    }
    __syncthreads();
  }

  // ---- Phase 4: out = O @ wp^T + bp. 64 tiles = 16 quads, f32 to HBM ----
  float* og = out + (size_t)blk * 64 * 256;
  for (int t = wv; t < 16; t += 8) {
    const int mq = t >> 3, nq = t & 7;
    const int fb = nq * 32;
    v8f acc[2][2] = {};
#pragma unroll 4
    for (int kk = 0; kk < 256; kk += 32) {
      const v16h a0 = load_a_frag(xs, 256, mq * 32, kk, lane);
      const v16h a1 = load_a_frag(xs, 256, mq * 32 + 16, kk, lane);
      const v16h b0 = load_bt_frag(wp16, 256, fb, kk, lane);
      const v16h b1 = load_bt_frag(wp16, 256, fb + 16, kk, lane);
      acc[0][0] = wmma_f16(a0, b0, acc[0][0]);
      acc[0][1] = wmma_f16(a0, b1, acc[0][1]);
      acc[1][0] = wmma_f16(a1, b0, acc[1][0]);
      acc[1][1] = wmma_f16(a1, b1, acc[1][1]);
    }
#pragma unroll
    for (int j = 0; j < 2; ++j) {
      const int n = fb + j * 16 + ln;
      const float bv = bp[n];
#pragma unroll
      for (int i = 0; i < 2; ++i)
#pragma unroll
        for (int r = 0; r < 8; ++r) {
          const int m = mq * 32 + i * 16 + r + hf * 8;
          og[m * 256 + n] = acc[i][j][r] + bv;
        }
    }
  }
}

// ---------------------------------------------------------------------------
extern "C" void kernel_launch(void* const* d_in, const int* in_sizes, int n_in,
                              void* d_out, int out_size, void* d_ws,
                              size_t ws_size, hipStream_t stream) {
  const float* x    = (const float*)d_in[0];
  const float* mask = (const float*)d_in[1];
  const float* wq   = (const float*)d_in[2];
  const float* bq   = (const float*)d_in[3];
  const float* wkv  = (const float*)d_in[4];
  const float* bkv  = (const float*)d_in[5];
  const float* wp   = (const float*)d_in[6];
  const float* bp   = (const float*)d_in[7];
  const float* rpb  = (const float*)d_in[8];
  const int*   rpi  = (const int*)d_in[9];

  char* ws = (char*)d_ws;
  _Float16* wq16  = (_Float16*)(ws);            // 256*256 f16 = 128 KiB
  _Float16* wkv16 = (_Float16*)(ws + 131072);   // 512*256 f16 = 256 KiB
  _Float16* wp16  = (_Float16*)(ws + 393216);   // 256*256 f16 = 128 KiB
  float*    biasg = (float*)(ws + 524288);      // 8*64*64 f32 = 128 KiB

  const int B  = in_sizes[0] / (64 * 256);      // 8192 windows
  const int nW = in_sizes[1] / (64 * 64);       // 4096 masks

  wa_prep_kernel<<<512, 256, 0, stream>>>(wq, wkv, wp, rpb, rpi, wq16, wkv16,
                                          wp16, biasg);
  wa_attn_kernel<<<B, 256, 0, stream>>>(x, mask, bq, bkv, bp, wq16, wkv16,
                                        wp16, biasg, (float*)d_out, nW);
}
